// AllostericGNN_25254407700536
// MI455X (gfx1250) — matrix operand order repaired and checked by reading, not compiled, not run on hardware
//
#include <hip/hip_runtime.h>
#include <hip/hip_bf16.h>
#include <math.h>

#define D_DIM   256
#define H_HEADS 8
#define DH_DIM  32
#define FFN_DIM 1024
#define L_LAYERS 2

typedef __attribute__((ext_vector_type(16))) __bf16 v16bf;
typedef __attribute__((ext_vector_type(8)))  float  v8f;
typedef unsigned short u16;
typedef unsigned int   u32;

__device__ __forceinline__ u16 f32_to_bf16(float f) {
    u32 u = __float_as_uint(f);
    u32 r = (u + 0x7FFFu + ((u >> 16) & 1u)) >> 16;   // round-to-nearest-even
    return (u16)r;
}
// order-preserving float<->uint encoding for atomicMax-based segment max
__device__ __forceinline__ u32 fenc(float f) {
    u32 u = __float_as_uint(f);
    return (u & 0x80000000u) ? ~u : (u | 0x80000000u);
}
__device__ __forceinline__ float fdec(u32 e) {
    return __uint_as_float((e & 0x80000000u) ? (e ^ 0x80000000u) : ~e);
}

// ---------------------------------------------------------------------------
// Tiled bf16 GEMM, f32 accumulation via v_wmma_f32_16x16x32_bf16.
// Block tile 64(M) x 128(N), BK=32, DOUBLE-BUFFERED LDS: stage tile k+1 while
// computing tile k -> one barrier per K step, VMEM latency overlapped with
// the WMMA pipe. 8 waves = 4(M) x 2(N); each wave owns a 16x64 strip of D
// (one A fragment reused across 4 WMMAs per K step).
// A staged row-major [64][40]; B staged transposed [128][40].
// Fragment packing follows CDNA5 ISA 7.12.2 16-bit A/B wave32 layouts.
// Epilogue: + bias, + optional residual, optional exact GELU; writes f32
// and/or bf16 (fusing the next GEMM's input conversion).
// ---------------------------------------------------------------------------
__global__ __launch_bounds__(256) void gemm_bf16_wmma(
    const u16* __restrict__ A, const u16* __restrict__ B,
    const float* __restrict__ bias, const float* __restrict__ residual,
    float* __restrict__ Cf, u16* __restrict__ Cb,
    int M, int N, int K, int gelu)
{
    __shared__ u16 As[2][64][40];    // 64 M-rows x 32 K (pad->40), x2 buffers
    __shared__ u16 Bt[2][128][40];   // 128 N-cols x 32 K (pad->40), x2 buffers

    const int tid  = threadIdx.x;
    const int wave = tid >> 5;
    const int lane = tid & 31;
    const int wm = wave >> 1;        // 0..3  (M wave tile)
    const int wn = wave & 1;         // 0..1  (N wave group of 64)
    const int lh = lane >> 4;        // lane half
    const int mr = lane & 15;

    const int m0 = blockIdx.y * 64;
    const int n0 = blockIdx.x * 128;

    const int arow = tid >> 2;          // 0..63
    const int akc  = (tid & 3) << 3;    // 0,8,16,24
    const int bkr  = tid >> 3;          // 0..31
    const int bnc  = (tid & 7) << 4;    // 0,16,...,112

    auto load_tile = [&](int k0, int buf) {
        {   // A tile: 16B per thread
            const u16* ap = A + (size_t)(m0 + arow) * K + (k0 + akc);
            uint4 av = *(const uint4*)ap;
            *(uint4*)&As[buf][arow][akc] = av;
            if (k0 + 32 < K) __builtin_prefetch(ap + 32, 0, 1);
        }
        {   // B tile: 32B per thread, stored transposed [n][k]
            const u16* bp = B + (size_t)(k0 + bkr) * N + (n0 + bnc);
            uint4 b0 = *(const uint4*)bp;
            uint4 b1 = *(const uint4*)(bp + 8);
            if (k0 + 32 < K) __builtin_prefetch(bp + (size_t)32 * N, 0, 1);
            u16 (*bt)[40] = Bt[buf];
            bt[bnc +  0][bkr] = (u16)(b0.x); bt[bnc +  1][bkr] = (u16)(b0.x >> 16);
            bt[bnc +  2][bkr] = (u16)(b0.y); bt[bnc +  3][bkr] = (u16)(b0.y >> 16);
            bt[bnc +  4][bkr] = (u16)(b0.z); bt[bnc +  5][bkr] = (u16)(b0.z >> 16);
            bt[bnc +  6][bkr] = (u16)(b0.w); bt[bnc +  7][bkr] = (u16)(b0.w >> 16);
            bt[bnc +  8][bkr] = (u16)(b1.x); bt[bnc +  9][bkr] = (u16)(b1.x >> 16);
            bt[bnc + 10][bkr] = (u16)(b1.y); bt[bnc + 11][bkr] = (u16)(b1.y >> 16);
            bt[bnc + 12][bkr] = (u16)(b1.z); bt[bnc + 13][bkr] = (u16)(b1.z >> 16);
            bt[bnc + 14][bkr] = (u16)(b1.w); bt[bnc + 15][bkr] = (u16)(b1.w >> 16);
        }
    };

    v8f acc0 = {}, acc1 = {}, acc2 = {}, acc3 = {};

    load_tile(0, 0);
    int cur = 0;
    for (int k0 = 0; k0 < K; k0 += 32) {
        __syncthreads();    // buf[cur] stores visible; prev reads of buf[cur^1] done
        if (k0 + 32 < K) load_tile(k0 + 32, cur ^ 1);

        union { v16bf v; u16 u[16]; } afrag;
        const int ar = wm * 16 + mr;
        // 16-bit A 16x32: halfs 0..7 -> K = lh*8 + i ; halfs 8..15 -> K = 16 + lh*8 + (i-8)
        #pragma unroll
        for (int i = 0; i < 8; ++i)  afrag.u[i] = As[cur][ar][lh * 8 + i];
        #pragma unroll
        for (int i = 8; i < 16; ++i) afrag.u[i] = As[cur][ar][8 + lh * 8 + i];

        // 16-bit B 32x16: lane = column, same K order as A (K = lh*8 + i, +8 for i >= 8)
        #pragma unroll
        for (int nn = 0; nn < 4; ++nn) {
            union { v16bf v; u16 u[16]; } bfrag;
            const int br = wn * 64 + nn * 16 + mr;
            #pragma unroll
            for (int i = 0; i < 16; ++i) bfrag.u[i] = Bt[cur][br][lh * 8 + i + ((i >> 3) << 3)];
            v8f a = (nn == 0) ? acc0 : (nn == 1) ? acc1 : (nn == 2) ? acc2 : acc3;
            a = __builtin_amdgcn_wmma_f32_16x16x32_bf16(
                false, afrag.v, false, bfrag.v, (short)0, a, false, false);
            if (nn == 0) acc0 = a; else if (nn == 1) acc1 = a;
            else if (nn == 2) acc2 = a; else acc3 = a;
        }
        cur ^= 1;
    }

    // C/D layout: VGPR r -> (M = lh*8 + r, N = mr)
    #pragma unroll
    for (int nn = 0; nn < 4; ++nn) {
        const v8f acc = (nn == 0) ? acc0 : (nn == 1) ? acc1 : (nn == 2) ? acc2 : acc3;
        const int gn = n0 + wn * 64 + nn * 16 + mr;
        const float bi = bias ? bias[gn] : 0.0f;
        #pragma unroll
        for (int r = 0; r < 8; ++r) {
            const int gm = m0 + wm * 16 + lh * 8 + r;
            const size_t idx = (size_t)gm * N + gn;
            float v = acc[r] + bi;
            if (residual) v += residual[idx];
            if (gelu) v = 0.5f * v * (1.0f + erff(v * 0.70710678118654752f));
            if (Cf) Cf[idx] = v;
            if (Cb) Cb[idx] = f32_to_bf16(v);
        }
    }
}

// ---------------------------------------------------------------------------
// Wave-per-row LayerNorm (D=256, 8 floats/lane), writes bf16 for next GEMM.
// ---------------------------------------------------------------------------
__global__ __launch_bounds__(256) void layernorm_to_bf16(
    const float* __restrict__ x, const float* __restrict__ g,
    const float* __restrict__ b, u16* __restrict__ out, int rows)
{
    const int wave = threadIdx.x >> 5;
    const int lane = threadIdx.x & 31;
    const int row  = blockIdx.x * 8 + wave;
    if (row >= rows) return;
    const float4* xp = (const float4*)(x + (size_t)row * D_DIM) + lane * 2;
    float4 a0 = xp[0], a1 = xp[1];
    float e[8] = {a0.x, a0.y, a0.z, a0.w, a1.x, a1.y, a1.z, a1.w};
    float s = 0.f;
    #pragma unroll
    for (int j = 0; j < 8; ++j) s += e[j];
    #pragma unroll
    for (int off = 16; off > 0; off >>= 1) s += __shfl_xor(s, off, 32);
    const float mu = s * (1.0f / D_DIM);
    float vs = 0.f;
    #pragma unroll
    for (int j = 0; j < 8; ++j) { float d = e[j] - mu; vs += d * d; }
    #pragma unroll
    for (int off = 16; off > 0; off >>= 1) vs += __shfl_xor(vs, off, 32);
    const float rs = rsqrtf(vs * (1.0f / D_DIM) + 1e-5f);
    u16* op = out + (size_t)row * D_DIM + lane * 8;
    const float* gp = g + lane * 8;
    const float* bp = b + lane * 8;
    #pragma unroll
    for (int j = 0; j < 8; ++j)
        op[j] = f32_to_bf16((e[j] - mu) * rs * gp[j] + bp[j]);
}

__global__ void f32_to_bf16_kernel(const float* __restrict__ in, u16* __restrict__ out, long n) {
    long i = (long)blockIdx.x * blockDim.x + threadIdx.x;
    if (i < n) out[i] = f32_to_bf16(in[i]);
}

__global__ void degree_kernel(const int* __restrict__ ei, int* __restrict__ deg, int E0) {
    int e = blockIdx.x * blockDim.x + threadIdx.x;
    if (e < E0) atomicAdd(&deg[ei[E0 + e]], 1);
}

// ---- edge-parallel attention (self edge e>=E0 -> node e-E0 appended) ------
__global__ void attn_score_kernel(const int* __restrict__ ei,
                                  const float* __restrict__ q, const float* __restrict__ k,
                                  float* __restrict__ sc, u32* __restrict__ mEnc,
                                  int E0, int Etot)
{
    int t = blockIdx.x * blockDim.x + threadIdx.x;
    if (t >= Etot * H_HEADS) return;
    int e = t >> 3, h = t & 7;
    int s = (e < E0) ? ei[e]      : (e - E0);
    int d = (e < E0) ? ei[E0 + e] : (e - E0);
    const float4* qp = (const float4*)(q + (size_t)d * D_DIM + h * DH_DIM);
    const float4* kp = (const float4*)(k + (size_t)s * D_DIM + h * DH_DIM);
    float acc = 0.f;
    #pragma unroll
    for (int j = 0; j < 8; ++j) {
        float4 a = qp[j], b = kp[j];
        acc += a.x * b.x + a.y * b.y + a.z * b.z + a.w * b.w;
    }
    acc *= 0.17677669529663689f;   // 1/sqrt(32)
    sc[t] = acc;
    atomicMax(&mEnc[d * H_HEADS + h], fenc(acc));
}

__global__ void attn_exp_kernel(const int* __restrict__ ei, float* __restrict__ sc,
                                const u32* __restrict__ mEnc, float* __restrict__ z,
                                int E0, int Etot)
{
    int t = blockIdx.x * blockDim.x + threadIdx.x;
    if (t >= Etot * H_HEADS) return;
    int e = t >> 3, h = t & 7;
    int d = (e < E0) ? ei[E0 + e] : (e - E0);
    float ex = expf(sc[t] - fdec(mEnc[d * H_HEADS + h]));
    sc[t] = ex;
    atomicAdd(&z[d * H_HEADS + h], ex);
}

__global__ void attn_agg_kernel(const int* __restrict__ ei, const float* __restrict__ sc,
                                const float* __restrict__ z, const float* __restrict__ v,
                                float* __restrict__ agg, int E0, int Etot)
{
    int t = blockIdx.x * blockDim.x + threadIdx.x;
    if (t >= Etot * H_HEADS) return;
    int e = t >> 3, h = t & 7;
    int s = (e < E0) ? ei[e]      : (e - E0);
    int d = (e < E0) ? ei[E0 + e] : (e - E0);
    float w = sc[t] / z[d * H_HEADS + h];
    const float4* vp = (const float4*)(v + (size_t)s * D_DIM + h * DH_DIM);
    float* ap = agg + (size_t)d * D_DIM + h * DH_DIM;
    #pragma unroll
    for (int j = 0; j < 8; ++j) {
        float4 vv = vp[j];
        atomicAdd(ap + j * 4 + 0, w * vv.x);
        atomicAdd(ap + j * 4 + 1, w * vv.y);
        atomicAdd(ap + j * 4 + 2, w * vv.z);
        atomicAdd(ap + j * 4 + 3, w * vv.w);
    }
}

__global__ void agg_mask_bf16_kernel(const float* __restrict__ agg, const int* __restrict__ deg,
                                     u16* __restrict__ out, long n) {
    long i = (long)blockIdx.x * blockDim.x + threadIdx.x;
    if (i >= n) return;
    int row = (int)(i >> 8);   // D_DIM == 256
    out[i] = f32_to_bf16(deg[row] > 0 ? agg[i] : 0.0f);
}

// ---------------------------------------------------------------------------
extern "C" void kernel_launch(void* const* d_in, const int* in_sizes, int n_in,
                              void* d_out, int out_size, void* d_ws, size_t ws_size,
                              hipStream_t stream)
{
    const float* x    = (const float*)d_in[0];
    const int*   ei   = (const int*)d_in[1];
    const float* w_in = (const float*)d_in[2];
    const float* b_in = (const float*)d_in[3];
    const float* ln1g = (const float*)d_in[4];
    const float* ln1b = (const float*)d_in[5];
    const float* ln2g = (const float*)d_in[6];
    const float* ln2b = (const float*)d_in[7];
    const float* wq = (const float*)d_in[8];  const float* bq = (const float*)d_in[9];
    const float* wk = (const float*)d_in[10]; const float* bkv = (const float*)d_in[11];
    const float* wv = (const float*)d_in[12]; const float* bvv = (const float*)d_in[13];
    const float* wo = (const float*)d_in[14]; const float* bo = (const float*)d_in[15];
    const float* w1 = (const float*)d_in[16]; const float* b1 = (const float*)d_in[17];
    const float* w2 = (const float*)d_in[18]; const float* b2 = (const float*)d_in[19];

    const int N    = in_sizes[0] / D_DIM;    // 32768
    const int E0   = in_sizes[1] / 2;        // 524288
    const int Etot = E0 + N;                 // + self edges

    char* p = (char*)d_ws;
    auto alloc = [&](size_t bytes) -> void* {
        void* r = (void*)p;
        p += (bytes + 255) & ~(size_t)255;
        return r;
    };

    u16*   xb    = (u16*)alloc((size_t)N * D_DIM * 2);
    u16*   xnb   = (u16*)alloc((size_t)N * D_DIM * 2);
    u16*   aggb  = (u16*)alloc((size_t)N * D_DIM * 2);
    u16*   ffnb  = (u16*)alloc((size_t)N * FFN_DIM * 2);
    float* h     = (float*)alloc((size_t)N * D_DIM * 4);
    float* qf    = (float*)alloc((size_t)N * D_DIM * 4);
    float* kf    = (float*)alloc((size_t)N * D_DIM * 4);
    float* vf    = (float*)alloc((size_t)N * D_DIM * 4);
    float* agg   = (float*)alloc((size_t)N * D_DIM * 4);
    float* sc    = (float*)alloc((size_t)Etot * H_HEADS * 4);
    u32*   mEnc  = (u32*)alloc((size_t)N * H_HEADS * 4);
    float* zf    = (float*)alloc((size_t)N * H_HEADS * 4);
    int*   deg   = (int*)alloc((size_t)N * 4);
    u16*   winb  = (u16*)alloc((size_t)D_DIM * D_DIM * 2);
    u16*   wqb   = (u16*)alloc((size_t)L_LAYERS * D_DIM * D_DIM * 2);
    u16*   wkb   = (u16*)alloc((size_t)L_LAYERS * D_DIM * D_DIM * 2);
    u16*   wvb   = (u16*)alloc((size_t)L_LAYERS * D_DIM * D_DIM * 2);
    u16*   wob   = (u16*)alloc((size_t)L_LAYERS * D_DIM * D_DIM * 2);
    u16*   w1b   = (u16*)alloc((size_t)L_LAYERS * D_DIM * FFN_DIM * 2);
    u16*   w2b   = (u16*)alloc((size_t)L_LAYERS * FFN_DIM * D_DIM * 2);

    auto conv = [&](const float* in, u16* out, long n) {
        f32_to_bf16_kernel<<<(int)((n + 255) / 256), 256, 0, stream>>>(in, out, n);
    };
    conv(x,    xb,   (long)N * D_DIM);
    conv(w_in, winb, (long)D_DIM * D_DIM);
    conv(wq,   wqb,  (long)L_LAYERS * D_DIM * D_DIM);
    conv(wk,   wkb,  (long)L_LAYERS * D_DIM * D_DIM);
    conv(wv,   wvb,  (long)L_LAYERS * D_DIM * D_DIM);
    conv(wo,   wob,  (long)L_LAYERS * D_DIM * D_DIM);
    conv(w1,   w1b,  (long)L_LAYERS * D_DIM * FFN_DIM);
    conv(w2,   w2b,  (long)L_LAYERS * FFN_DIM * D_DIM);

    hipMemsetAsync(deg, 0, (size_t)N * 4, stream);
    degree_kernel<<<(E0 + 255) / 256, 256, 0, stream>>>(ei, deg, E0);

    auto gemm = [&](const u16* A, const u16* B, const float* bias, const float* res,
                    float* Cf, u16* Cb, int M, int Nn, int K, int gelu) {
        dim3 grid(Nn / 128, M / 64);
        gemm_bf16_wmma<<<grid, 256, 0, stream>>>(A, B, bias, res, Cf, Cb, M, Nn, K, gelu);
    };

    // h = x @ w_in + b_in
    gemm(xb, winb, b_in, nullptr, h, nullptr, N, D_DIM, D_DIM, 0);

    const int eth = Etot * H_HEADS;
    for (int l = 0; l < L_LAYERS; ++l) {
        layernorm_to_bf16<<<(N + 7) / 8, 256, 0, stream>>>(h, ln1g + l * D_DIM, ln1b + l * D_DIM, xnb, N);
        gemm(xnb, wqb + (size_t)l * D_DIM * D_DIM, bq  + l * D_DIM, nullptr, qf, nullptr, N, D_DIM, D_DIM, 0);
        gemm(xnb, wkb + (size_t)l * D_DIM * D_DIM, bkv + l * D_DIM, nullptr, kf, nullptr, N, D_DIM, D_DIM, 0);
        gemm(xnb, wvb + (size_t)l * D_DIM * D_DIM, bvv + l * D_DIM, nullptr, vf, nullptr, N, D_DIM, D_DIM, 0);

        hipMemsetAsync(mEnc, 0, (size_t)N * H_HEADS * 4, stream);  // fenc identity (< any finite key)
        hipMemsetAsync(zf,   0, (size_t)N * H_HEADS * 4, stream);
        hipMemsetAsync(agg,  0, (size_t)N * D_DIM * 4, stream);
        attn_score_kernel<<<(eth + 255) / 256, 256, 0, stream>>>(ei, qf, kf, sc, mEnc, E0, Etot);
        attn_exp_kernel  <<<(eth + 255) / 256, 256, 0, stream>>>(ei, sc, mEnc, zf, E0, Etot);
        attn_agg_kernel  <<<(eth + 255) / 256, 256, 0, stream>>>(ei, sc, zf, vf, agg, E0, Etot);
        agg_mask_bf16_kernel<<<(int)(((long)N * D_DIM + 255) / 256), 256, 0, stream>>>(agg, deg, aggb, (long)N * D_DIM);

        // h += masked_agg @ wo + bo
        gemm(aggb, wob + (size_t)l * D_DIM * D_DIM, bo + l * D_DIM, h, h, nullptr, N, D_DIM, D_DIM, 0);

        layernorm_to_bf16<<<(N + 7) / 8, 256, 0, stream>>>(h, ln2g + l * D_DIM, ln2b + l * D_DIM, xnb, N);
        // ffn = gelu(xn2 @ w1 + b1)  -> bf16 only
        gemm(xnb, w1b + (size_t)l * D_DIM * FFN_DIM, b1 + l * FFN_DIM, nullptr, nullptr, ffnb, N, FFN_DIM, D_DIM, 1);
        // h += ffn @ w2 + b2  (last layer writes straight to d_out)
        float* outp = (l == L_LAYERS - 1) ? (float*)d_out : h;
        gemm(ffnb, w2b + (size_t)l * FFN_DIM * D_DIM, b2 + l * D_DIM, h, outp, nullptr, N, D_DIM, FFN_DIM, 0);
    }
    (void)n_in; (void)out_size; (void)ws_size;
}
